// AffineAttentionNN_53171695125356
// MI455X (gfx1250) — matrix operand order repaired and checked by reading, not compiled, not run on hardware
//
#include <hip/hip_runtime.h>
#include <hip/hip_bf16.h>
#include <math.h>

// Problem sizes (fixed by the reference)
#define NROWS 16384
#define DIN   128
#define DOUT  128
#define NC    128

// Main-kernel tiling
#define WGT   512          // threads per workgroup (16 waves)
#define MT    4            // M-tiles (of 16 rows) per wave
#define APITCH 136         // padded float pitch of the LDS a-tile (32B-aligned rows)

typedef __attribute__((ext_vector_type(16))) __bf16 v16bf;
typedef __attribute__((ext_vector_type(8)))  __bf16 v8bf;
typedef __attribute__((ext_vector_type(8)))  float  v8f;

// ---------------------------------------------------------------------------
// Prep 1: WvT[c][p][g] = bf16(Wv[c][g][p])   (4 MB bf16, lives in L2)
// Transposed so WMMA B-fragments (consecutive K per lane) are contiguous.
// ---------------------------------------------------------------------------
__global__ __launch_bounds__(256) void prep_wvt_kernel(
    const float* __restrict__ wv, __bf16* __restrict__ wvt) {
  size_t i = (size_t)blockIdx.x * 256 + threadIdx.x;   // dest index, 2^21 total
  int g = (int)(i & 127);
  int p = (int)((i >> 7) & 127);
  int c = (int)(i >> 14);
  wvt[i] = (__bf16)wv[((size_t)c << 14) + ((size_t)g << 7) + p];
}

// Prep 2: x -> bf16
__global__ __launch_bounds__(256) void prep_xb_kernel(
    const float* __restrict__ x, __bf16* __restrict__ xb) {
  size_t i = (size_t)blockIdx.x * 256 + threadIdx.x;
  xb[i] = (__bf16)x[i];
}

// Prep 3: cn[c] = sum_g ctrs[c,g]^2 * s[g]
__global__ __launch_bounds__(128) void prep_cn_kernel(
    const float* __restrict__ ctrs, const float* __restrict__ s,
    float* __restrict__ cn) {
  __shared__ float ss[DIN];
  int c = threadIdx.x;
  ss[c] = s[c];
  __syncthreads();
  const float* cr = ctrs + (size_t)c * DIN;
  float acc = 0.f;
  #pragma unroll 8
  for (int g = 0; g < DIN; ++g) acc = __builtin_fmaf(cr[g] * cr[g], ss[g], acc);
  cn[c] = acc;
}

// ---------------------------------------------------------------------------
// Assignments: a[n,c] = softmax_c( -( ||x||_s^2 - 2 (x*s)@ctrs^T + cn[c] ) )
// ~0.5 GFLOP total: VALU is the right tool here.
// ---------------------------------------------------------------------------
__global__ __launch_bounds__(128) void assign_kernel(
    const float* __restrict__ x, const float* __restrict__ ctrs,
    const float* __restrict__ s, const float* __restrict__ cn,
    float* __restrict__ a) {
  __shared__ float xs[DIN];
  __shared__ float red[NC];
  const int n = blockIdx.x;
  const int c = threadIdx.x;

  float xv = x[(size_t)n * DIN + c];
  float sv = s[c];
  xs[c]  = xv * sv;
  red[c] = xv * xv * sv;
  __syncthreads();

  for (int off = 64; off > 0; off >>= 1) {
    if (c < off) red[c] += red[c + off];
    __syncthreads();
  }
  float xxs = red[0];
  __syncthreads();

  const float* cr = ctrs + (size_t)c * DIN;
  float dot = 0.f;
  #pragma unroll 8
  for (int g = 0; g < DIN; ++g) dot = __builtin_fmaf(xs[g], cr[g], dot);

  float logit = -(xxs - 2.f * dot + cn[c]);

  red[c] = logit;
  __syncthreads();
  for (int off = 64; off > 0; off >>= 1) {
    if (c < off) red[c] = fmaxf(red[c], red[c + off]);
    __syncthreads();
  }
  float mx = red[0];
  __syncthreads();
  float ex = __expf(logit - mx);
  red[c] = ex;
  __syncthreads();
  for (int off = 64; off > 0; off >>= 1) {
    if (c < off) red[c] += red[c + off];
    __syncthreads();
  }
  float inv = 1.f / red[0];
  a[(size_t)n * NC + c] = ex * inv;
}

// ---------------------------------------------------------------------------
// Main einsum via bf16 WMMA with async-staged LDS B panels:
//   out[n,p] = sum_c a[n,c] * ( x[n,:] @ Wv[c,:,p] + Ov[c,p] )
//
// Workgroup = 512 threads (16 waves) = 128 rows x 128 outputs.
//   wave w: p-tile pt = w&7 (16 cols), row group mg = w>>3 (4 M-tiles = 64 rows)
// Per center c:
//   - panel WvT[c] (32 KB) staged once per WG via global_load_async_to_lds_b128
//     (double buffered, overlapped with WMMA of the previous panel)
//   - all 4 B fragments + row scales hoisted into registers first (one load
//     clause), then 16 WMMAs run as 4 independent chains -> minimal dscnt stalls
//   - t-tiles scaled by a[n,c] with Ov[c,p] folded in
// L2 traffic: 128 WGs x 4 MB = 512 MB (8x less than per-wave streaming).
// ---------------------------------------------------------------------------
__global__ __launch_bounds__(WGT) void moe_wmma_kernel(
    const __bf16* __restrict__ xb,    // [NROWS, DIN]  bf16
    const __bf16* __restrict__ wvt,   // [NC, DOUT, DIN] bf16 (g<->p transposed)
    const float*  __restrict__ a,     // [NROWS, NC]   f32
    const float*  __restrict__ ov,    // [NC, DOUT]    f32
    float* __restrict__ out) {        // [NROWS, DOUT] f32
  extern __shared__ char smem[];
  __bf16* ldsB  = (__bf16*)smem;                 // [2][128*128] bf16 = 64 KB
  float*  lds_a = (float*)(smem + 65536);        // [128][APITCH] f32 ~= 68 KB

  const int m0   = blockIdx.x * 128;
  const int tid  = threadIdx.x;
  const int w    = tid >> 5;
  const int lane = tid & 31;
  const int col  = lane & 15;          // A-frag row within tile / B&D column
  const int half = lane >> 4;          // lane half selects 16-wide K half
  const int pt   = w & 7;
  const int mg   = w >> 3;             // 0 or 1
  const int p    = pt * 16 + col;
  const int rb   = m0 + mg * 64;       // wave's first row

  // Stage a[m0:m0+128, :] transposed into LDS: lds_a[c][m] (coalesced reads).
  for (int i = tid; i < 128 * NC; i += WGT) {
    int m = i >> 7;
    int c = i & 127;
    lds_a[c * APITCH + m] = a[(size_t)(m0 + m) * NC + c];
  }

  // Preload c-invariant x A-fragments (16x32 bf16, contiguous K order matching B):
  // lane m = col; elems 0..15 -> K = ks*32 + half*16 + e
  v16bf xf[MT][4];
  #pragma unroll
  for (int mt = 0; mt < MT; ++mt) {
    const __bf16* xrow = xb + (size_t)(rb + mt * 16 + col) * DIN;
    #pragma unroll
    for (int ks = 0; ks < 4; ++ks) {
      xf[mt][ks] = *(const v16bf*)(xrow + ks * 32 + half * 16);
    }
  }

  // Prologue: async-stage B panel for c=0 into buffer 0.
  // 512 threads x 4 instr x 16 B = 32 KB, lane-coalesced.
  {
    const char* g  = (const char*)wvt;
    uint32_t    l0 = (uint32_t)(uintptr_t)ldsB;
    #pragma unroll
    for (int j = 0; j < 4; ++j) {
      uint32_t    lo_ = l0 + (uint32_t)(j * 8192 + tid * 16);
      const char* gp  = g + j * 8192 + tid * 16;
      asm volatile("global_load_async_to_lds_b128 %0, %1, off"
                   :: "v"(lo_), "v"(gp) : "memory");
    }
  }

  v8f acc[MT] = {};

  for (int c = 0; c < NC; ++c) {
    const int cur = c & 1;

    // Stage next panel into the other buffer, then wait for the current one.
    if (c + 1 < NC) {
      const char* g  = (const char*)(wvt + (size_t)(c + 1) * (DOUT * DIN));
      uint32_t    l0 = (uint32_t)(uintptr_t)(ldsB + (cur ^ 1) * (128 * 128));
      #pragma unroll
      for (int j = 0; j < 4; ++j) {
        uint32_t    lo_ = l0 + (uint32_t)(j * 8192 + tid * 16);
        const char* gp  = g + j * 8192 + tid * 16;
        asm volatile("global_load_async_to_lds_b128 %0, %1, off"
                     :: "v"(lo_), "v"(gp) : "memory");
      }
      // Async loads complete in order: <=4 outstanding => panel c is in LDS.
      asm volatile("s_wait_asynccnt 0x4" ::: "memory");
    } else {
      asm volatile("s_wait_asynccnt 0x0" ::: "memory");
    }
    __syncthreads();   // panel `cur` visible to all waves

    // Hoist ALL loads for this center first (one clause-able batch):
    //  - 4 B fragments (8x ds_load_b128)
    //  - 4 v8f row-scale vectors (8x ds_load_b128)
    //  - Ov scalar
    const __bf16* bp = ldsB + (size_t)cur * (128 * 128) + (size_t)p * DIN + half * 16;
    v16bf bfr[4];
    #pragma unroll
    for (int ks = 0; ks < 4; ++ks)
      bfr[ks] = *(const v16bf*)(bp + ks * 32);

    v8f av[MT];
    #pragma unroll
    for (int mt = 0; mt < MT; ++mt)
      av[mt] = *(const v8f*)&lds_a[c * APITCH + mg * 64 + mt * 16 + half * 8];

    float ovv = ov[(size_t)c * DOUT + p];

    // 16 WMMAs as 4 independent accumulation chains (one per M-tile).
    v8f t[MT] = {};
    #pragma unroll
    for (int ks = 0; ks < 4; ++ks) {
      #pragma unroll
      for (int mt = 0; mt < MT; ++mt)
        t[mt] = __builtin_amdgcn_wmma_f32_16x16x32_bf16(
            false, xf[mt][ks], false, bfr[ks], (short)0, t[mt], false, false);
    }

    #pragma unroll
    for (int mt = 0; mt < MT; ++mt) {
      #pragma unroll
      for (int r = 0; r < 8; ++r)
        acc[mt][r] = __builtin_fmaf(av[mt][r], t[mt][r] + ovv, acc[mt][r]);
    }
    __syncthreads();   // all waves done with `cur` before it is restaged
  }

  // C/D layout: VGPR r holds (M = r + half*8, N = col).
  #pragma unroll
  for (int mt = 0; mt < MT; ++mt) {
    float* orow = out + (size_t)(rb + mt * 16 + half * 8) * DOUT + p;
    #pragma unroll
    for (int r = 0; r < 8; ++r)
      orow[(size_t)r * DOUT] = acc[mt][r];
  }
}

// ---------------------------------------------------------------------------
extern "C" void kernel_launch(void* const* d_in, const int* in_sizes, int n_in,
                              void* d_out, int out_size, void* d_ws,
                              size_t ws_size, hipStream_t stream) {
  const float* x    = (const float*)d_in[0];  // [16384,128]
  const float* ctrs = (const float*)d_in[1];  // [128,128]
  const float* Wv   = (const float*)d_in[2];  // [128,128,128]
  const float* Ov   = (const float*)d_in[3];  // [128,128]
  const float* s    = (const float*)d_in[4];  // [128]
  float* out = (float*)d_out;                 // [16384,128]

  // Workspace layout (~16.1 MB):
  char* ws = (char*)d_ws;
  __bf16* wvt = (__bf16*)ws;                          // 4 MB
  __bf16* xb  = (__bf16*)(ws + (4u << 20));           // 4 MB
  float*  cn  = (float*)(ws + (8u << 20));            // 512 B
  float*  a   = (float*)(ws + (8u << 20) + 4096);     // 8 MB

  prep_wvt_kernel<<<(NC * DOUT * DIN) / 256, 256, 0, stream>>>(Wv, wvt);
  prep_xb_kernel<<<(NROWS * DIN) / 256, 256, 0, stream>>>(x, xb);
  prep_cn_kernel<<<1, 128, 0, stream>>>(ctrs, s, cn);
  assign_kernel<<<NROWS, 128, 0, stream>>>(x, ctrs, s, cn, a);

  const size_t smem_bytes = 65536 + (size_t)NC * APITCH * sizeof(float);
  moe_wmma_kernel<<<NROWS / 128, WGT, smem_bytes, stream>>>(xb, wvt, a, Ov, out);
}
